// MultiHeadAttentionBlock_8306466750694
// MI455X (gfx1250) — matrix operand order repaired and checked by reading, not compiled, not run on hardware
//
#include <hip/hip_runtime.h>
#include <hip/hip_bf16.h>
#include <cstdint>
#include <cstddef>

typedef __bf16 bf16;
typedef __attribute__((ext_vector_type(16))) __bf16 v16bf;
typedef __attribute__((ext_vector_type(8)))  __bf16 v8bf;
typedef __attribute__((ext_vector_type(8)))  float  v8f;
typedef __attribute__((ext_vector_type(4)))  float  v4f;

#define B_SZ    2
#define L_SEQ   2048
#define DMODEL  1024
#define NHEADS  16
#define DKH     64

// ---------- fragment helpers ----------
__device__ __forceinline__ v8bf ld8(const bf16* p) { return *(const v8bf*)p; }

__device__ __forceinline__ v16bf pack16(v8bf lo, v8bf hi) {
  v16bf r;
#pragma unroll
  for (int i = 0; i < 8; ++i) { r[i] = lo[i]; r[i + 8] = hi[i]; }
  return r;
}

__device__ __forceinline__ v8f wmma_bf16(v16bf a, v16bf b, v8f c) {
  // D(f32 16x16) = A(bf16 16x32) * B(bf16 32x16) + C
  return __builtin_amdgcn_wmma_f32_16x16x32_bf16(false, a, false, b,
                                                 (short)0, c, false, false);
}

// ---------- elementwise f32 -> bf16 (activation pre-convert) ----------
__global__ void __launch_bounds__(256)
cvt_f32_to_bf16(const float* __restrict__ in, bf16* __restrict__ out) {
  const size_t i = ((size_t)blockIdx.x * blockDim.x + threadIdx.x) * 8;
  v4f a = *(const v4f*)(in + i);
  v4f b = *(const v4f*)(in + i + 4);
  v8bf r;
#pragma unroll
  for (int j = 0; j < 4; ++j) { r[j] = (bf16)a[j]; r[j + 4] = (bf16)b[j]; }
  *(v8bf*)(out + i) = r;
}

// ---------- key-padding mask -> bf16 0/1 vector ----------
__global__ void __launch_bounds__(256)
mask_to_bf16(const int* __restrict__ m, bf16* __restrict__ mb) {
  const int i = blockIdx.x * blockDim.x + threadIdx.x;
  mb[i] = (bf16)((m[i] > 0) ? 1.0f : 0.0f);
}

// ---------- weight transpose: W[k][n] f32 -> Wt[n][k] bf16 ----------
__global__ void __launch_bounds__(256)
transpose_to_bf16(const float* __restrict__ W, bf16* __restrict__ Wt) {
  __shared__ float tile[16][17];
  const int n0 = blockIdx.x * 16, k0 = blockIdx.y * 16;
  const int tx = threadIdx.x, ty = threadIdx.y;
  tile[ty][tx] = W[(size_t)(k0 + ty) * DMODEL + (n0 + tx)];
  __syncthreads();
  Wt[(size_t)(n0 + ty) * DMODEL + (k0 + tx)] = (bf16)tile[tx][ty];
}

// ---------- GEMM: out = (X(M x K) * W(K x N) + bias) * oscale, via Wt[n][k] --
// wave computes 16 rows x 64 cols (A fragment reused by 4 WMMAs per k-step)
// OUT_MODE: 0 = f32 row-major, 1 = bf16 row-major,
//           2 = bf16 (B,H,dk,L) V-transpose with key-padding rows zeroed
template <int OUT_MODE>
__global__ void __launch_bounds__(128)
gemm_wmma(const bf16* __restrict__ X, const bf16* __restrict__ Wt,
          const float* __restrict__ bias, void* __restrict__ outp,
          float oscale, const int* __restrict__ vmask) {
  const int N = DMODEL, K = DMODEL;
  const int wave = threadIdx.x >> 5, lane = threadIdx.x & 31;
  const int laneN = lane & 15, hi = lane >> 4;
  const int m0 = (blockIdx.x >> 2) * 16;          // 256 M-tiles
  const int n0 = (blockIdx.x & 3) * 256 + wave * 64;
  const int kb8  = hi * 8;   // unused; A also loads contiguous k via koff
  const int koff = hi * 16;  // B-fragment half-wave K offset

  const bf16* xrow = X + (size_t)(m0 + laneN) * K;
  const bf16* w0 = Wt + (size_t)(n0 + laneN) * K;
  const bf16* w1 = w0 + (size_t)16 * K;
  const bf16* w2 = w0 + (size_t)32 * K;
  const bf16* w3 = w0 + (size_t)48 * K;

  v8f acc0 = {}, acc1 = {}, acc2 = {}, acc3 = {};
#pragma unroll 2
  for (int k0 = 0; k0 < K; k0 += 32) {
    __builtin_prefetch(xrow + ((k0 + 256) & (K - 1)), 0, 1);  // global_prefetch
    v16bf a  = pack16(ld8(xrow + k0 + koff), ld8(xrow + k0 + koff + 8));
    v16bf b0 = pack16(ld8(w0 + k0 + koff), ld8(w0 + k0 + koff + 8));
    v16bf b1 = pack16(ld8(w1 + k0 + koff), ld8(w1 + k0 + koff + 8));
    v16bf b2 = pack16(ld8(w2 + k0 + koff), ld8(w2 + k0 + koff + 8));
    v16bf b3 = pack16(ld8(w3 + k0 + koff), ld8(w3 + k0 + koff + 8));
    acc0 = wmma_bf16(a, b0, acc0);
    acc1 = wmma_bf16(a, b1, acc1);
    acc2 = wmma_bf16(a, b2, acc2);
    acc3 = wmma_bf16(a, b3, acc3);
  }

  // mode 2: fold key-padding mask into V rows (tokens m0+8*hi .. +7)
  float mf[8];
  if (OUT_MODE == 2) {
    const int bb = m0 >> 11, t0 = (m0 & (L_SEQ - 1)) + 8 * hi;
    const int4 mA = *(const int4*)(vmask + bb * L_SEQ + t0);
    const int4 mB = *(const int4*)(vmask + bb * L_SEQ + t0 + 4);
    mf[0] = mA.x > 0 ? 1.f : 0.f; mf[1] = mA.y > 0 ? 1.f : 0.f;
    mf[2] = mA.z > 0 ? 1.f : 0.f; mf[3] = mA.w > 0 ? 1.f : 0.f;
    mf[4] = mB.x > 0 ? 1.f : 0.f; mf[5] = mB.y > 0 ? 1.f : 0.f;
    mf[6] = mB.z > 0 ? 1.f : 0.f; mf[7] = mB.w > 0 ? 1.f : 0.f;
  }

#pragma unroll
  for (int nt = 0; nt < 4; ++nt) {
    const v8f acc = (nt == 0) ? acc0 : (nt == 1) ? acc1 : (nt == 2) ? acc2 : acc3;
    const int n = n0 + nt * 16 + laneN;
    const float bv = bias[n];
    if (OUT_MODE == 2) {
      // (B,H,dk,L): 8 consecutive tokens per lane are contiguous -> b128 store
      const int bb = m0 >> 11, hh = n >> 6, d = n & (DKH - 1);
      v8bf pkt;
#pragma unroll
      for (int r = 0; r < 8; ++r) pkt[r] = (bf16)((acc[r] + bv) * mf[r]);
      bf16* dst = (bf16*)outp +
                  ((size_t)((bb * NHEADS + hh) * DKH) + d) * L_SEQ +
                  (m0 & (L_SEQ - 1)) + 8 * hi;
      *(v8bf*)dst = pkt;
    } else {
#pragma unroll
      for (int r = 0; r < 8; ++r) {
        const int row = m0 + r + 8 * hi;
        const float val = (acc[r] + bv) * oscale;
        if (OUT_MODE == 0) ((float*)outp)[(size_t)row * N + n] = val;
        else               ((bf16*)outp)[(size_t)row * N + n] = (bf16)val;
      }
    }
  }
}

// ---------- one 64-key attention tile ----------------------------------------
// CAUSAL instantiation only used for the single diagonal tile per wave.
// Padding mask handled by mask-weighted rowsum B operand + pre-masked V,
// so the non-causal (hot) path has no compares/selects at all.
template <bool CAUSAL>
__device__ __forceinline__ void attn_tile(
    int kb, int q0, int laneN, int hi, int kb8,
    const bf16* __restrict__ kp, const bf16* __restrict__ vcol,
    const bf16* __restrict__ mbt,   // maskbf + b*L + koff + kb
    v16bf qa0, v16bf qa1, bf16* P, float* mrow, float* lrow, v8f* o) {
  // scores: S[16 x 64] as 4 column-group fragments (Q pre-scaled by 1/8)
  v8f s[4];
#pragma unroll
  for (int g = 0; g < 4; ++g) {
    const bf16* kg = kp + (size_t)g * 16 * DMODEL;
    const v16bf kA = pack16(ld8(kg),      ld8(kg + 8));
    const v16bf kB = pack16(ld8(kg + 32), ld8(kg + 40));
    v8f sg = {};
    sg = wmma_bf16(qa0, kA, sg);
    sg = wmma_bf16(qa1, kB, sg);
    s[g] = sg;
  }

  float alphaA[8];
#pragma unroll
  for (int r = 0; r < 8; ++r) {
    const int qi = q0 + r + 8 * hi;
    // mask-free running max: softmax(o/l) invariant to the normalizer; e <= 1
    float mx = fmaxf(fmaxf(s[0][r], s[1][r]), fmaxf(s[2][r], s[3][r]));
    mx = fmaxf(mx, __shfl_xor(mx, 1));
    mx = fmaxf(mx, __shfl_xor(mx, 2));
    mx = fmaxf(mx, __shfl_xor(mx, 4));
    mx = fmaxf(mx, __shfl_xor(mx, 8));
    const float mnew = fmaxf(mrow[r], mx);
    const float alpha = __expf(mrow[r] - mnew);
    alphaA[r] = alpha;
    mrow[r] = mnew;
    const int row = r + 8 * hi;
#pragma unroll
    for (int g = 0; g < 4; ++g) {
      float e = __expf(s[g][r] - mnew);
      if (CAUSAL) e = ((kb + g * 16 + laneN) <= qi) ? e : 0.f;
      P[row * 72 + g * 16 + laneN] = (bf16)e;
    }
#pragma unroll
    for (int nt = 0; nt < 4; ++nt) o[nt][r] *= alpha;
  }

  // P tile: C-layout -> A-layout via per-wave LDS (in-order DS, no barrier)
  const v16bf pa0 = pack16(*(const v8bf*)&P[laneN * 72 + 2 * kb8],
                           *(const v8bf*)&P[laneN * 72 + 2 * kb8 + 8]);
  const v16bf pa1 = pack16(*(const v8bf*)&P[laneN * 72 + 32 + 2 * kb8],
                           *(const v8bf*)&P[laneN * 72 + 40 + 2 * kb8]);

  // denominator: rowsum weighted by bf16 0/1 key mask (B depends on k only)
  const v16bf mf0 = pack16(ld8(mbt),      ld8(mbt + 8));
  const v16bf mf1 = pack16(ld8(mbt + 32), ld8(mbt + 40));
  v8f z = {};
  v8f rsum = wmma_bf16(pa0, mf0, z);
  rsum = wmma_bf16(pa1, mf1, rsum);
#pragma unroll
  for (int r = 0; r < 8; ++r) lrow[r] = lrow[r] * alphaA[r] + rsum[r];

  // PV: B-fragments from pre-masked V-transpose layout (contiguous along keys)
#pragma unroll
  for (int nt = 0; nt < 4; ++nt) {
    const bf16* vg = vcol + (size_t)nt * 16 * L_SEQ;
    const v16bf vA = pack16(ld8(vg),      ld8(vg + 8));
    const v16bf vB = pack16(ld8(vg + 32), ld8(vg + 40));
    o[nt] = wmma_bf16(pa0, vA, o[nt]);
    o[nt] = wmma_bf16(pa1, vB, o[nt]);
  }
}

// ---------- flash attention: one wave per (b, h, 16-query tile) --------------
__global__ void __launch_bounds__(128)
attn_flash(const bf16* __restrict__ Qp,     // (B,L,D) bf16, pre-scaled by 0.125
           const bf16* __restrict__ Kp,     // (B,L,D) bf16
           const bf16* __restrict__ Vt,     // (B,H,dk,L) bf16, pad rows zeroed
           const bf16* __restrict__ maskbf, // (B,L) bf16 0/1
           bf16* __restrict__ Xo) {         // (B,L,D) bf16
  __shared__ __align__(16) bf16 Pbuf[4][16 * 72];  // 16x64 tile, row stride 72
  const int wave = threadIdx.x >> 5, lane = threadIdx.x & 31;
  const int laneN = lane & 15, hi = lane >> 4;
  const int wg = blockIdx.x * 4 + wave;
  const int qtPerBH = L_SEQ / 16;
  const int b   = wg / (NHEADS * qtPerBH);
  const int rem = wg % (NHEADS * qtPerBH);
  const int h   = rem / qtPerBH;
  const int q0  = (rem % qtPerBH) * 16;

  const int kb8 = hi * 8, koff = hi * 16;

  // Q A-fragments (dk 0..31 and 32..63)
  const bf16* qrow = Qp + (size_t)(b * L_SEQ + q0 + laneN) * DMODEL + h * DKH;
  const v16bf qa0 = pack16(ld8(qrow + koff),      ld8(qrow + koff + 8));
  const v16bf qa1 = pack16(ld8(qrow + 32 + koff), ld8(qrow + 40 + koff));

  float mrow[8], lrow[8];
  v8f o[4] = {};
#pragma unroll
  for (int r = 0; r < 8; ++r) { mrow[r] = -1e30f; lrow[r] = 0.f; }

  bf16* P = &Pbuf[wave][0];
  // loop-carried per-lane pointers; in-tile offsets become load immediates
  const bf16* kp   = Kp + (size_t)(b * L_SEQ + laneN) * DMODEL + h * DKH + koff;
  const bf16* vcol = Vt + (size_t)((b * NHEADS + h) * DKH + laneN) * L_SEQ + koff;
  const bf16* mb   = maskbf + b * L_SEQ + koff;

  // exactly one diagonal 64-key tile per 16-row q-tile; the rest are full
  const int nfull = q0 >> 6;
  int kb = 0;
  for (int t = 0; t < nfull; ++t, kb += 64) {
    __builtin_prefetch(kp + (size_t)64 * DMODEL, 0, 1);
    __builtin_prefetch(vcol + 64, 0, 1);
    attn_tile<false>(kb, q0, laneN, hi, kb8, kp, vcol, mb + kb,
                     qa0, qa1, P, mrow, lrow, o);
    kp   += (size_t)64 * DMODEL;
    vcol += 64;
  }
  attn_tile<true>(kb, q0, laneN, hi, kb8, kp, vcol, mb + kb,
                  qa0, qa1, P, mrow, lrow, o);

#pragma unroll
  for (int r = 0; r < 8; ++r) {
    const float inv = (lrow[r] > 0.f) ? 1.f / lrow[r] : 0.f;
    const int tok = q0 + r + 8 * hi;
#pragma unroll
    for (int nt = 0; nt < 4; ++nt) {
      const int col = h * DKH + nt * 16 + laneN;
      Xo[(size_t)(b * L_SEQ + tok) * DMODEL + col] = (bf16)(o[nt][r] * inv);
    }
  }
}

// ---------- launch ----------
extern "C" void kernel_launch(void* const* d_in, const int* in_sizes, int n_in,
                              void* d_out, int out_size, void* d_ws, size_t ws_size,
                              hipStream_t stream) {
  (void)in_sizes; (void)n_in; (void)out_size; (void)ws_size;
  const float* q    = (const float*)d_in[0];
  const float* k    = (const float*)d_in[1];
  const float* v    = (const float*)d_in[2];
  const int*   mask = (const int*)d_in[3];
  const float* Wq = (const float*)d_in[4];  const float* bq = (const float*)d_in[5];
  const float* Wk = (const float*)d_in[6];  const float* bk = (const float*)d_in[7];
  const float* Wv = (const float*)d_in[8];  const float* bv = (const float*)d_in[9];
  const float* Wo = (const float*)d_in[10]; const float* bo = (const float*)d_in[11];
  float* out = (float*)d_out;

  const size_t WT  = (size_t)DMODEL * DMODEL;       // 1M elems (2MB bf16)
  const size_t TOK = (size_t)B_SZ * L_SEQ * DMODEL; // 4M elems (8MB bf16)

  bf16* p = (bf16*)d_ws;
  bf16* WtQ = p; p += WT;
  bf16* WtK = p; p += WT;
  bf16* WtV = p; p += WT;
  bf16* WtO = p; p += WT;
  bf16* Xq  = p; p += TOK;
  bf16* Xk  = p; p += TOK;
  bf16* Xv  = p; p += TOK;
  bf16* Qp  = p; p += TOK;
  bf16* Kp  = p; p += TOK;
  bf16* Vtr = p; p += TOK;
  bf16* Xo  = p; p += TOK;
  bf16* Mbf = p; p += (size_t)B_SZ * L_SEQ;   // total 64 MB + 8KB

  // activation pre-convert to bf16 (removes v_cvt from GEMM hot loop)
  {
    dim3 cg((unsigned)(TOK / (256 * 8)));
    cvt_f32_to_bf16<<<cg, 256, 0, stream>>>(q, Xq);
    cvt_f32_to_bf16<<<cg, 256, 0, stream>>>(k, Xk);
    cvt_f32_to_bf16<<<cg, 256, 0, stream>>>(v, Xv);
  }
  mask_to_bf16<<<(B_SZ * L_SEQ) / 256, 256, 0, stream>>>(mask, Mbf);

  dim3 tb(16, 16), tg(DMODEL / 16, DMODEL / 16);
  transpose_to_bf16<<<tg, tb, 0, stream>>>(Wq, WtQ);
  transpose_to_bf16<<<tg, tb, 0, stream>>>(Wk, WtK);
  transpose_to_bf16<<<tg, tb, 0, stream>>>(Wv, WtV);
  transpose_to_bf16<<<tg, tb, 0, stream>>>(Wo, WtO);

  const int M = B_SZ * L_SEQ;
  dim3 gg((M / 16) * (DMODEL / 256));  // 1024 blocks: 4 waves x 16x64 each
  gemm_wmma<1><<<gg, 128, 0, stream>>>(Xq, WtQ, bq, Qp, 0.125f, nullptr);
  gemm_wmma<1><<<gg, 128, 0, stream>>>(Xk, WtK, bk, Kp, 1.0f, nullptr);
  gemm_wmma<2><<<gg, 128, 0, stream>>>(Xv, WtV, bv, Vtr, 1.0f, mask);

  const int nwaves = B_SZ * NHEADS * (L_SEQ / 16);  // 4096
  attn_flash<<<nwaves / 4, 128, 0, stream>>>(Qp, Kp, Vtr, Mbf, Xo);

  gemm_wmma<0><<<gg, 128, 0, stream>>>(Xo, WtO, bo, out, 1.0f, nullptr);
}
